// SelfAttention3D_77335181132119
// MI455X (gfx1250) — matrix operand order repaired and checked by reading, not compiled, not run on hardware
//
#include <hip/hip_runtime.h>
#include <math.h>

// ---------------- problem constants ----------------
#define B_      2
#define C_      256
#define N_      4096            // D*H*W = 16^3
#define GROUPS_ 8
#define CPG     32              // channels per group
#define NH      4
#define HD      32
#define INNER   128             // NH*HD
#define QKVR    384             // 3*INNER
#define EPS_    1e-5f

typedef float  v8f   __attribute__((ext_vector_type(8)));
typedef __bf16 v8bf  __attribute__((ext_vector_type(8)));
typedef __bf16 v16bf __attribute__((ext_vector_type(16)));

// bf16 WMMA, f32 accumulate: D(16x16 f32) = A(16x32 bf16) * B(32x16 bf16) + C
static __device__ __forceinline__ v8f wmma_bf16(v16bf a, v16bf b, v8f c) {
    return __builtin_amdgcn_wmma_f32_16x16x32_bf16(false, a, false, b, (short)0, c,
                                                   false, false);
}
static __device__ __forceinline__ v16bf cat16(v8bf lo, v8bf hi) {
    return __builtin_shufflevector(lo, hi, 0,1,2,3,4,5,6,7,8,9,10,11,12,13,14,15);
}

// ---------------- 1) GroupNorm statistics (f32) ----------------
__global__ void gn_stats_kernel(const float* __restrict__ x,
                                float* __restrict__ stats) {
    const int bg = blockIdx.x;               // b*GROUPS_ + g
    const int b  = bg / GROUPS_;
    const int g  = bg % GROUPS_;
    const float* xp = x + ((size_t)b * C_ + (size_t)g * CPG) * N_;
    const int tot = CPG * N_;

    float s = 0.f, s2 = 0.f;
    for (int i = threadIdx.x; i < tot; i += blockDim.x) {
        float v = xp[i];
        s += v; s2 += v * v;
    }
    __shared__ float sh0[256], sh1[256];
    sh0[threadIdx.x] = s; sh1[threadIdx.x] = s2;
    __syncthreads();
    for (int off = 128; off > 0; off >>= 1) {
        if ((int)threadIdx.x < off) {
            sh0[threadIdx.x] += sh0[threadIdx.x + off];
            sh1[threadIdx.x] += sh1[threadIdx.x + off];
        }
        __syncthreads();
    }
    if (threadIdx.x == 0) {
        float inv  = 1.0f / (float)tot;
        float mean = sh0[0] * inv;
        float var  = sh1[0] * inv - mean * mean;
        stats[bg * 2 + 0] = mean;
        stats[bg * 2 + 1] = rsqrtf(var + EPS_);
    }
}

// ---------------- 2) weights -> bf16 (once) ----------------
__global__ void cvt_weights_kernel(const float* __restrict__ qkv_w,
                                   const float* __restrict__ proj_w,
                                   __bf16* __restrict__ qkv_wb,
                                   __bf16* __restrict__ proj_wb) {
    int i = blockIdx.x * 256 + threadIdx.x;        // covers 98304 + 32768
    if (i < QKVR * C_) qkv_wb[i] = (__bf16)qkv_w[i];
    else               proj_wb[i - QKVR * C_] = (__bf16)proj_w[i - QKVR * C_];
}

// ---------------- 3) GroupNorm apply + transpose to xnT[n][c] bf16 --------
__global__ void gn_apply_t_kernel(const float* __restrict__ x,
                                  const float* __restrict__ w,
                                  const float* __restrict__ bias,
                                  const float* __restrict__ stats,
                                  __bf16* __restrict__ xnT) {
    int idx = blockIdx.x * 256 + threadIdx.x;      // over B*N*C (c fastest)
    int c = idx & (C_ - 1);
    int n = (idx >> 8) & (N_ - 1);
    int b = idx >> 20;
    int g = c / CPG;
    float mean = stats[(b * GROUPS_ + g) * 2 + 0];
    float rstd = stats[(b * GROUPS_ + g) * 2 + 1];
    float v = x[((size_t)b * C_ + c) * N_ + n];
    xnT[idx] = (__bf16)((v - mean) * rstd * w[c] + bias[c]);
}

// ---------------- 4) QKV GEMM (bf16 WMMA), epilogue into q/k/v layouts ----
__global__ void qkv_gemm_kernel(const __bf16* __restrict__ qkv_wb,
                                const __bf16* __restrict__ xnT,
                                __bf16* __restrict__ qT,   // [b,h][n][d]
                                __bf16* __restrict__ kT,   // [b,h][n][d]
                                __bf16* __restrict__ vmat) // [b,h][d][n]
{
    const int wave = blockIdx.x * (blockDim.x >> 5) + (threadIdx.x >> 5);
    const int lane = threadIdx.x & 31;
    const int half = lane >> 4;
    const int li   = lane & 15;
    int t  = wave;
    int nt = t & 255; t >>= 8;                 // 256 n-tiles
    int mt = t % 24;  t /= 24;                 // 24 m-tiles
    int b  = t;

    const int n = nt * 16 + li;
    const int m = mt * 16 + li;
    const __bf16* Arow = qkv_wb + (size_t)m * C_;
    const __bf16* Brow = xnT + ((size_t)b * N_ + n) * C_;

    v8f acc = {0.f,0.f,0.f,0.f,0.f,0.f,0.f,0.f};
    #pragma unroll
    for (int k = 0; k < C_; k += 32) {
        v16bf af = *(const v16bf*)(Arow + k + half * 16);
        v16bf bf = *(const v16bf*)(Brow + k + half * 16);
        acc = wmma_bf16(af, bf, acc);
    }

    const int o0 = mt * 16;
    const int s  = o0 / INNER;                 // 0=q 1=k 2=v (uniform per wave)
    const int h  = (o0 % INNER) / HD;
    const int d0 = o0 % HD;                    // 0 or 16
    if (s < 2) {
        __bf16* dst = (s == 0 ? qT : kT)
                    + ((size_t)(b * NH + h) * N_ + n) * HD + d0 + half * 8;
        v8bf pk;
        #pragma unroll
        for (int v = 0; v < 8; ++v) pk[v] = (__bf16)acc[v];
        *(v8bf*)dst = pk;
    } else {
        #pragma unroll
        for (int v = 0; v < 8; ++v) {
            int d = d0 + v + half * 8;
            vmat[((size_t)(b * NH + h) * HD + d) * N_ + n] = (__bf16)acc[v];
        }
    }
}

// ---- online-softmax update for one i-tile over a 32-j block ----
static __device__ __forceinline__ float softmax_step(v8f& sA, v8f& sB,
                                                     float& m_i, float& l_i,
                                                     v8bf& pA, v8bf& pB) {
    const float scale = 0.17677669529663687f;     // HD^-0.5
    float mx = -1e30f;
    #pragma unroll
    for (int v = 0; v < 8; ++v) {
        sA[v] *= scale; sB[v] *= scale;
        mx = fmaxf(mx, fmaxf(sA[v], sB[v]));
    }
    mx = fmaxf(mx, __shfl_xor(mx, 16, 32));
    float m_new = fmaxf(m_i, mx);
    float alpha = __expf(m_i - m_new);
    float psum  = 0.f;
    #pragma unroll
    for (int v = 0; v < 8; ++v) {
        float eA = __expf(sA[v] - m_new); psum += eA; pA[v] = (__bf16)eA;
        float eB = __expf(sB[v] - m_new); psum += eB; pB[v] = (__bf16)eB;
    }
    psum += __shfl_xor(psum, 16, 32);
    l_i = l_i * alpha + psum;
    m_i = m_new;
    return alpha;
}

// ---------------- 5) Flash attention: 2 i-tiles per wave ------------------
// K/V fragments are i-invariant: each loaded fragment feeds 2 WMMAs.
__global__ void flash_attn_kernel(const __bf16* __restrict__ qT,
                                  const __bf16* __restrict__ kT,
                                  const __bf16* __restrict__ vmat,
                                  __bf16* __restrict__ aoT)  // [b][n][128]
{
    const int wib  = threadIdx.x >> 5;
    const int wave = blockIdx.x * 4 + wib;
    const int lane = threadIdx.x & 31;
    const int half = lane >> 4;
    const int li   = lane & 15;
    int t   = wave;
    int itp = t & 127; t >>= 7;        // i-pair (32 columns)
    int h   = t & 3;   t >>= 2;
    int b   = t;
    const int bh = b * NH + h;

    const __bf16* kbase = kT   + (size_t)bh * N_ * HD;
    const __bf16* vbase = vmat + (size_t)bh * HD * N_;
    const int i_col0 = itp * 32 + li;
    const int i_col1 = i_col0 + 16;

    // Q B-fragments: loop invariant (K = d = 32 -> one fragment per i-tile)
    v16bf qb0 = *(const v16bf*)(qT + ((size_t)bh * N_ + i_col0) * HD + half * 16);
    v16bf qb1 = *(const v16bf*)(qT + ((size_t)bh * N_ + i_col1) * HD + half * 16);

    float m0 = -1e30f, l0 = 0.f, m1 = -1e30f, l1 = 0.f;
    v8f acc00 = {0.f,0.f,0.f,0.f,0.f,0.f,0.f,0.f};   // tile0, d 0..15
    v8f acc01 = {0.f,0.f,0.f,0.f,0.f,0.f,0.f,0.f};   // tile0, d 16..31
    v8f acc10 = {0.f,0.f,0.f,0.f,0.f,0.f,0.f,0.f};   // tile1, d 0..15
    v8f acc11 = {0.f,0.f,0.f,0.f,0.f,0.f,0.f,0.f};   // tile1, d 16..31

    __shared__ __bf16 pbuf[4][2][16][32];            // per-wave P^T staging (8KB)

    for (int j0 = 0; j0 < N_; j0 += 32) {
        if (j0 + 32 < N_) {   // prefetch next K/V rows -> global_prefetch_b8
            __builtin_prefetch(kbase + (size_t)(j0 + 32 + li) * HD, 0, 0);
            __builtin_prefetch(vbase + (size_t)li * N_ + j0 + 32, 0, 0);
        }

        // shared K fragments (j rows 0..15 and 16..31 of the block)
        const __bf16* kr0 = kbase + (size_t)(j0 + li) * HD;
        const __bf16* kr1 = kbase + (size_t)(j0 + 16 + li) * HD;
        v16bf ka0 = *(const v16bf*)(kr0 + half * 16);
        v16bf ka1 = *(const v16bf*)(kr1 + half * 16);

        // S^T tiles: one WMMA each (K covers all 32 d); fragments reused 2x
        v8f z = {0.f,0.f,0.f,0.f,0.f,0.f,0.f,0.f};
        v8f s0A = wmma_bf16(ka0, qb0, z);
        v8f s0B = wmma_bf16(ka1, qb0, z);
        v8f s1A = wmma_bf16(ka0, qb1, z);
        v8f s1B = wmma_bf16(ka1, qb1, z);

        v8bf p0A, p0B, p1A, p1B;
        float a0 = softmax_step(s0A, s0B, m0, l0, p0A, p0B);
        float a1 = softmax_step(s1A, s1B, m1, l1, p1A, p1B);
        #pragma unroll
        for (int v = 0; v < 8; ++v) {
            acc00[v] *= a0; acc01[v] *= a0;
            acc10[v] *= a1; acc11[v] *= a1;
        }

        // P^T: C layout -> B fragment via per-wave LDS (in-order DS pipe)
        *(v8bf*)&pbuf[wib][0][li][half * 8]      = p0A;
        *(v8bf*)&pbuf[wib][0][li][16 + half * 8] = p0B;
        *(v8bf*)&pbuf[wib][1][li][half * 8]      = p1A;
        *(v8bf*)&pbuf[wib][1][li][16 + half * 8] = p1B;
        v16bf pb0 = *(const v16bf*)&pbuf[wib][0][li][half * 16];
        v16bf pb1 = *(const v16bf*)&pbuf[wib][1][li][half * 16];

        // shared V fragments (d rows 0..15 and 16..31); each reused 2x
        const __bf16* vr0 = vbase + (size_t)li * N_ + j0;
        const __bf16* vr1 = vbase + (size_t)(16 + li) * N_ + j0;
        v16bf va0 = *(const v16bf*)(vr0 + half * 16);
        v16bf va1 = *(const v16bf*)(vr1 + half * 16);
        acc00 = wmma_bf16(va0, pb0, acc00);
        acc01 = wmma_bf16(va1, pb0, acc01);
        acc10 = wmma_bf16(va0, pb1, acc10);
        acc11 = wmma_bf16(va1, pb1, acc11);
    }

    // epilogue: aoT[n][inner] bf16, rows are consecutive d -> packed stores
    {
        const float inv_l = 1.0f / l0;
        __bf16* orow = aoT + ((size_t)b * N_ + i_col0) * INNER + h * HD;
        v8bf oA, oB;
        #pragma unroll
        for (int v = 0; v < 8; ++v) {
            oA[v] = (__bf16)(acc00[v] * inv_l);
            oB[v] = (__bf16)(acc01[v] * inv_l);
        }
        *(v8bf*)(orow + half * 8)      = oA;
        *(v8bf*)(orow + 16 + half * 8) = oB;
    }
    {
        const float inv_l = 1.0f / l1;
        __bf16* orow = aoT + ((size_t)b * N_ + i_col1) * INNER + h * HD;
        v8bf oA, oB;
        #pragma unroll
        for (int v = 0; v < 8; ++v) {
            oA[v] = (__bf16)(acc10[v] * inv_l);
            oB[v] = (__bf16)(acc11[v] * inv_l);
        }
        *(v8bf*)(orow + half * 8)      = oA;
        *(v8bf*)(orow + 16 + half * 8) = oB;
    }
}

// ---------------- 6) Proj GEMM (bf16 WMMA) + bias + residual (f32) --------
__global__ void proj_gemm_kernel(const __bf16* __restrict__ proj_wb,
                                 const __bf16* __restrict__ aoT,
                                 const float* __restrict__ proj_b,
                                 const float* __restrict__ x,
                                 float* __restrict__ out) {
    const int wave = blockIdx.x * (blockDim.x >> 5) + (threadIdx.x >> 5);
    const int lane = threadIdx.x & 31;
    const int half = lane >> 4;
    const int li   = lane & 15;
    int t  = wave;
    int nt = t & 255; t >>= 8;                 // 256 n-tiles
    int mt = t & 15;  t >>= 4;                 // 16 m-tiles
    int b  = t;

    const int m = mt * 16 + li;
    const int n = nt * 16 + li;
    const __bf16* Arow = proj_wb + (size_t)m * INNER;
    const __bf16* Brow = aoT + ((size_t)b * N_ + n) * INNER;

    v8f acc = {0.f,0.f,0.f,0.f,0.f,0.f,0.f,0.f};
    #pragma unroll
    for (int k = 0; k < INNER; k += 32) {
        v16bf af = *(const v16bf*)(Arow + k + half * 16);
        v16bf bf = *(const v16bf*)(Brow + k + half * 16);
        acc = wmma_bf16(af, bf, acc);
    }
    #pragma unroll
    for (int v = 0; v < 8; ++v) {
        int row = mt * 16 + v + half * 8;
        size_t idx = ((size_t)b * C_ + row) * N_ + n;
        out[idx] = acc[v] + proj_b[row] + x[idx];
    }
}

// ---------------- launcher ----------------
extern "C" void kernel_launch(void* const* d_in, const int* in_sizes, int n_in,
                              void* d_out, int out_size, void* d_ws, size_t ws_size,
                              hipStream_t stream) {
    (void)in_sizes; (void)n_in; (void)out_size; (void)ws_size;
    const float* x      = (const float*)d_in[0];
    const float* norm_w = (const float*)d_in[1];
    const float* norm_b = (const float*)d_in[2];
    const float* qkv_w  = (const float*)d_in[3];
    const float* proj_w = (const float*)d_in[4];
    const float* proj_b = (const float*)d_in[5];
    float* out = (float*)d_out;

    // workspace layout (all offsets 256B aligned)
    char* ws = (char*)d_ws;
    float*  stats   = (float*) (ws + 0);                     //   256 B
    __bf16* qkv_wb  = (__bf16*)(ws + 256);                   // 196608
    __bf16* proj_wb = (__bf16*)(ws + 196864);                //  65536
    __bf16* xnT     = (__bf16*)(ws + 262400);                // 4 MiB  [b][n][c]
    __bf16* qT      = (__bf16*)(ws + 4456704);               // 2 MiB  [b,h][n][d]
    __bf16* kT      = (__bf16*)(ws + 6553856);               // 2 MiB  [b,h][n][d]
    __bf16* vmat    = (__bf16*)(ws + 8651008);               // 2 MiB  [b,h][d][n]
    __bf16* aoT     = (__bf16*)(ws + 10748160);              // 2 MiB  [b][n][128]

    gn_stats_kernel<<<B_ * GROUPS_, 256, 0, stream>>>(x, stats);

    cvt_weights_kernel<<<(QKVR * C_ + C_ * INNER) / 256, 256, 0, stream>>>(
        qkv_w, proj_w, qkv_wb, proj_wb);

    gn_apply_t_kernel<<<(B_ * N_ * C_) / 256, 256, 0, stream>>>(
        x, norm_w, norm_b, stats, xnT);

    // 2 * 24 * 256 = 12288 waves, 8 waves/block
    qkv_gemm_kernel<<<12288 / 8, 256, 0, stream>>>(qkv_wb, xnT, qT, kT, vmat);

    // 2 * 4 * 128 = 1024 waves (2 i-tiles each), 4 waves/block
    flash_attn_kernel<<<1024 / 4, 128, 0, stream>>>(qT, kT, vmat, aoT);

    // 2 * 16 * 256 = 8192 waves, 8 waves/block
    proj_gemm_kernel<<<8192 / 8, 256, 0, stream>>>(proj_wb, aoT, proj_b, x, out);
}
